// InnerAttention_11751030522443
// MI455X (gfx1250) — compile-verified, matrix operand order checked by reading, not run on hardware
//
#include <hip/hip_runtime.h>
#include <hip/hip_bf16.h>

typedef __bf16 bf16;
typedef __attribute__((ext_vector_type(16))) __bf16 v16bf;
typedef __attribute__((ext_vector_type(8)))  __bf16 v8bf;
typedef __attribute__((ext_vector_type(8)))  float  v8f;
typedef __attribute__((ext_vector_type(4)))  unsigned int v4u;
typedef __attribute__((ext_vector_type(8)))  int v8i;
typedef __attribute__((ext_vector_type(4)))  int v4i;

#define NTOK   1024
#define NH     8
#define HDIM   64
#define CDIM   512
#define BATCH  8
#define CONVK  15
#define QSCALE 0.125f   // 64^-0.5

// ---- workspace layout (bytes) ----
static constexpr size_t OFF_XB  = 0;                                  // x  bf16 [8192,512]
static constexpr size_t OFF_WQB = OFF_XB  + (size_t)8192*512*2;       // w_qkv bf16 [1536,512]
static constexpr size_t OFF_WPB = OFF_WQB + (size_t)1536*512*2;       // w_proj bf16 [512,512]
static constexpr size_t OFF_QB  = OFF_WPB + (size_t)512*512*2;        // q*scale bf16 [B,H,N,HD]
static constexpr size_t OFF_KB  = OFF_QB  + (size_t)BATCH*NH*NTOK*HDIM*2;
static constexpr size_t OFF_VT  = OFF_KB  + (size_t)BATCH*NH*NTOK*HDIM*2; // v bf16 [B,H,HD,N]
static constexpr size_t OFF_AO  = OFF_VT  + (size_t)BATCH*NH*NTOK*HDIM*2; // attn out bf16 [B,N,C]

// ---------------------------------------------------------------------------
// Fragment helpers. CDNA5 16-bit A/B 16x32 layout (ISA 7.12.2):
//   lanes 0-15  : row M = lane,    K = {0..7} in v0-3, {16..23} in v4-7
//   lanes 16-31 : row M = lane-16, K = {8..15},        {24..31}
// => per lane: two contiguous 8xbf16 (16B) chunks at kbase+sel*8 and +16.
// ---------------------------------------------------------------------------
__device__ __forceinline__ v16bf load_frag(const bf16* base, int row, int ld,
                                           int kbase, int lane) {
  const bf16* p = base + (size_t)row * (size_t)ld + kbase + ((lane >> 4) << 3);
  v8bf lo = *(const v8bf*)p;
  v8bf hi = *(const v8bf*)(p + 16);
  v16bf r;
#pragma unroll
  for (int i = 0; i < 8; ++i) { r[i] = lo[i]; r[i + 8] = hi[i]; }
  return r;
}

__device__ __forceinline__ v8f wmma_bf16(v16bf a, v16bf b, v8f c) {
  // (neg_a, A, neg_b, B, c_mod, C, reuse_a, reuse_b)
  return __builtin_amdgcn_wmma_f32_16x16x32_bf16(false, a, false, b, (short)0, c,
                                                 false, false);
}

// ---------------------------------------------------------------------------
// Kernel 0: fp32 -> bf16 conversion of x, w_qkv, w_proj
// ---------------------------------------------------------------------------
__global__ void cvt_bf16_kernel(const float* __restrict__ x,
                                const float* __restrict__ wq,
                                const float* __restrict__ wp,
                                bf16* __restrict__ xb, bf16* __restrict__ wqb,
                                bf16* __restrict__ wpb) {
  int idx = blockIdx.x * blockDim.x + threadIdx.x;
  const int NX = 8192 * 512;        // 4194304
  const int NQ = 1536 * 512;        // 786432
  const int NP = 512 * 512;         // 262144
  if (idx < NX) {
    xb[idx] = (bf16)x[idx];
  } else if (idx < NX + NQ) {
    int j = idx - NX;
    wqb[j] = (bf16)wq[j];
  } else if (idx < NX + NQ + NP) {
    int j = idx - NX - NQ;
    wpb[j] = (bf16)wp[j];
  }
}

// ---------------------------------------------------------------------------
// Kernel 1: QKV GEMM  qkv[i,j] = sum_c x[i,c]*w_qkv[j,c] + b_qkv[j]
// One 16x16 output tile per wave, K=512 (16 bf16 WMMAs).
// Scatter into q(scaled)/k [B,H,N,HD] and vT [B,H,HD,N] bf16.
// ---------------------------------------------------------------------------
__global__ void qkv_gemm_kernel(const bf16* __restrict__ xb,
                                const bf16* __restrict__ wqb,
                                const float* __restrict__ b_qkv,
                                bf16* __restrict__ qb, bf16* __restrict__ kb,
                                bf16* __restrict__ vT) {
  int wave = threadIdx.x >> 5, lane = threadIdx.x & 31;
  int tile = blockIdx.x * 8 + wave;       // 512*96 tiles total
  int it = tile / 96, jt = tile % 96;
  int arow = it * 16 + (lane & 15);
  int brow = jt * 16 + (lane & 15);
  v8f acc = {};
  for (int k = 0; k < 512; k += 32) {
    __builtin_prefetch(xb + (size_t)arow * 512 + k + 64, 0, 1);
    v16bf a = load_frag(xb, arow, 512, k, lane);
    v16bf b = load_frag(wqb, brow, 512, k, lane);
    acc = wmma_bf16(a, b, acc);
  }
  int j = jt * 16 + (lane & 15);
  float bj = b_qkv[j];
  int mh = (lane >> 4) << 3;
#pragma unroll
  for (int r = 0; r < 8; ++r) {
    int i = it * 16 + r + mh;             // global row (b*1024+n)
    float val = acc[r] + bj;
    int bidx = i >> 10, n = i & 1023;
    if (j < 512) {            // Q (pre-scaled)
      int h = j >> 6, d = j & 63;
      qb[(((size_t)(bidx * 8 + h)) * 1024 + n) * 64 + d] = (bf16)(val * QSCALE);
    } else if (j < 1024) {    // K
      int jj = j - 512; int h = jj >> 6, d = jj & 63;
      kb[(((size_t)(bidx * 8 + h)) * 1024 + n) * 64 + d] = (bf16)val;
    } else {                  // V, transposed
      int jj = j - 1024; int h = jj >> 6, d = jj & 63;
      vT[(((size_t)(bidx * 8 + h)) * 64 + d) * 1024 + n] = (bf16)val;
    }
  }
}

// ---------------------------------------------------------------------------
// Kernel 2: fused attention per (b, h, 32-row block).
// LDS: S fp32 [48][1024] (32 rows + 8 halo each side for the 15-tap conv),
//      P bf16 [32][1024] (unnormalized exp), Q tile bf16 [48][64] (TDM-staged),
//      reductions.
// ---------------------------------------------------------------------------
#define SMEM_S_BYTES  (48 * 1024 * 4)
#define SMEM_P_BYTES  (32 * 1024 * 2)
#define SMEM_QS_OFF   (SMEM_S_BYTES + SMEM_P_BYTES)
#define SMEM_QS_BYTES (48 * 64 * 2)
#define SMEM_RED_OFF  (SMEM_QS_OFF + SMEM_QS_BYTES)
#define SMEM_TOTAL    (SMEM_RED_OFF + (256 + 256 + 32) * 4)

__global__ void attn_kernel(const bf16* __restrict__ qb,
                            const bf16* __restrict__ kb,
                            const bf16* __restrict__ vT,
                            const float* __restrict__ rel,
                            const float* __restrict__ pe_w,
                            const float* __restrict__ pe_b,
                            bf16* __restrict__ AO) {
  extern __shared__ char smem[];
  float* S      = (float*)smem;                         // [48][1024]
  bf16*  P      = (bf16*)(smem + SMEM_S_BYTES);         // [32][1024]
  bf16*  Qs     = (bf16*)(smem + SMEM_QS_OFF);          // [48][64]
  float* redmax = (float*)(smem + SMEM_RED_OFF);        // [256]
  float* redsum = redmax + 256;                         // [256]
  float* rowinv = redsum + 256;                         // [32]

  int rb = blockIdx.x & 31;
  int bh = blockIdx.x >> 5;
  int b = bh >> 3, h = bh & 7;
  int r0 = rb * 32;
  int tid = threadIdx.x;
  int wave = tid >> 5, lane = tid & 31;

  const bf16* qbase = qb + (size_t)bh * 1024 * 64;
  const bf16* kbase = kb + (size_t)bh * 1024 * 64;
  const bf16* vbase = vT + (size_t)bh * 64 * 1024;

  // ---- Stage Q tile (48x64 = query rows + conv halo) into LDS via TDM.
  // Pre-zero the region: covers top-edge halo rows the descriptor cannot
  // address (negative rows); bottom-edge OOB rows are zero-filled by TDM.
  {
    unsigned int* qz = (unsigned int*)Qs;
    for (int i = tid; i < SMEM_QS_BYTES / 4; i += 256) qz[i] = 0u;
  }
  __syncthreads();
  if (wave == 0) {
    int start = r0 - 8;
    int skip = start < 0 ? -start : 0;     // rows clipped at the top edge
    int row0 = start + skip;
    int rows = 48 - skip;                  // tile_dim1
    int dim1 = 1024 - row0;                // rows to tensor end (zero past it)
    unsigned long long ga =
        (unsigned long long)(const void*)(qbase + (size_t)row0 * 64);
    unsigned int lds = (unsigned int)(SMEM_QS_OFF + skip * 64 * 2);
    // D# group 0: count=1 | lds_addr | global_addr[56:0] | type=2 (ISA 8.3)
    v4u g0 = {1u, lds, (unsigned int)(ga & 0xffffffffu),
              (unsigned int)((ga >> 32) & 0x01ffffffu) | (2u << 30)};
    // D# group 1 (ISA 8.4): data_size=1 (2B); tensor_dim0=64; tensor_dim1;
    // tile_dim0=64; tile_dim1=rows; tensor_dim0_stride=64.
    v8i g1 = {(int)(1u << 16),
              (int)(64u << 16),
              (int)(((unsigned)dim1 & 0xffffu) << 16),
              (int)(64u << 16),
              rows,
              64,
              0, 0};
    v4i gz = {0, 0, 0, 0};
    v8i gz8 = {0, 0, 0, 0, 0, 0, 0, 0};
    __builtin_amdgcn_tensor_load_to_lds(g0, g1, gz, gz, gz8, 0);
    __builtin_amdgcn_s_wait_tensorcnt(0);
  }
  __syncthreads();

  // ---- Phase 1: scores S[48][1024] = (q*scale) @ k^T; A from LDS Q tile.
  for (int t = wave; t < 3 * 64; t += 8) {
    int tr = t >> 6, tc = t & 63;
    int lr = tr * 16 + (lane & 15);        // local Q row, halo rows are zero
    int m = tc * 16 + (lane & 15);
    v8f acc = {};
#pragma unroll
    for (int k = 0; k < 64; k += 32) {
      v16bf a = load_frag(Qs, lr, 64, k, lane);
      v16bf bb = load_frag(kbase, m, 64, k, lane);
      acc = wmma_bf16(a, bb, acc);
    }
    int mh = (lane >> 4) << 3;
    int col = tc * 16 + (lane & 15);
#pragma unroll
    for (int r = 0; r < 8; ++r) S[(tr * 16 + r + mh) * 1024 + col] = acc[r];
  }
  __syncthreads();

  // ---- Phase 2: depthwise conv over query axis + rel-pos bias + softmax.
  // 8 threads per output row, 128 cols each; two passes (max, then exp/sum).
  {
    int o = tid >> 3;          // output row in block, 0..31
    int slot = tid & 7;
    int c0 = slot * 128;
    int n = r0 + o;
    int ny = n >> 5, nx = n & 31;
    float w[CONVK];
#pragma unroll
    for (int k = 0; k < CONVK; ++k) w[k] = pe_w[h * CONVK + k];
    float peb = pe_b[h];
    const float* Srow = S + (o + 8) * 1024;

    float mx = -3.4e38f;
    for (int m = c0; m < c0 + 128; ++m) {
      float a = Srow[m] + peb;
#pragma unroll
      for (int k = 0; k < CONVK; ++k) a += w[k] * S[(o + 1 + k) * 1024 + m];
      int my = m >> 5, mcx = m & 31;
      a += rel[((ny - my + 31) * 63 + (nx - mcx + 31)) * 8 + h];
      mx = fmaxf(mx, a);
    }
    redmax[tid] = mx;
    __syncthreads();
    float rowm = redmax[o * 8];
#pragma unroll
    for (int s = 1; s < 8; ++s) rowm = fmaxf(rowm, redmax[o * 8 + s]);

    float sum = 0.f;
    for (int m = c0; m < c0 + 128; ++m) {
      float a = Srow[m] + peb;
#pragma unroll
      for (int k = 0; k < CONVK; ++k) a += w[k] * S[(o + 1 + k) * 1024 + m];
      int my = m >> 5, mcx = m & 31;
      a += rel[((ny - my + 31) * 63 + (nx - mcx + 31)) * 8 + h];
      float e = __expf(a - rowm);
      sum += e;
      P[o * 1024 + m] = (bf16)e;   // unnormalized; 1/sum applied after PV
    }
    redsum[tid] = sum;
    __syncthreads();
    float rs = redsum[o * 8];
#pragma unroll
    for (int s = 1; s < 8; ++s) rs += redsum[o * 8 + s];
    if (slot == 0) rowinv[o] = 1.0f / rs;
  }
  __syncthreads();

  // ---- Phase 3: out = (P @ V) * rowinv. One 16x16 tile per wave (2x4 tiles),
  // K=1024 -> 32 bf16 WMMAs. A from LDS, B from global vT (contiguous).
  {
    int tr = wave >> 2, tc = wave & 3;
    int prow = tr * 16 + (lane & 15);
    int drow = tc * 16 + (lane & 15);
    v8f acc = {};
    for (int k = 0; k < 1024; k += 32) {
      v16bf a = load_frag(P, prow, 1024, k, lane);
      v16bf bb = load_frag(vbase, drow, 1024, k, lane);
      acc = wmma_bf16(a, bb, acc);
    }
    int mh = (lane >> 4) << 3;
    int d = tc * 16 + (lane & 15);
#pragma unroll
    for (int r = 0; r < 8; ++r) {
      int ro = tr * 16 + r + mh;
      float val = acc[r] * rowinv[ro];
      int gn = r0 + ro;
      AO[((size_t)(b * 1024 + gn)) * 512 + h * 64 + d] = (bf16)val;
    }
  }
}

// ---------------------------------------------------------------------------
// Kernel 3: output projection  out[i,j] = sum_c AO[i,c]*w_proj[j,c] + b_proj[j]
// ---------------------------------------------------------------------------
__global__ void proj_gemm_kernel(const bf16* __restrict__ AO,
                                 const bf16* __restrict__ wpb,
                                 const float* __restrict__ b_proj,
                                 float* __restrict__ out) {
  int wave = threadIdx.x >> 5, lane = threadIdx.x & 31;
  int tile = blockIdx.x * 8 + wave;       // 512*32 tiles
  int it = tile >> 5, jt = tile & 31;
  int arow = it * 16 + (lane & 15);
  int brow = jt * 16 + (lane & 15);
  v8f acc = {};
  for (int k = 0; k < 512; k += 32) {
    __builtin_prefetch(AO + (size_t)arow * 512 + k + 64, 0, 1);
    v16bf a = load_frag(AO, arow, 512, k, lane);
    v16bf b = load_frag(wpb, brow, 512, k, lane);
    acc = wmma_bf16(a, b, acc);
  }
  int j = jt * 16 + (lane & 15);
  float bj = b_proj[j];
  int mh = (lane >> 4) << 3;
#pragma unroll
  for (int r = 0; r < 8; ++r) {
    int i = it * 16 + r + mh;
    out[(size_t)i * 512 + j] = acc[r] + bj;
  }
}

// ---------------------------------------------------------------------------
extern "C" void kernel_launch(void* const* d_in, const int* in_sizes, int n_in,
                              void* d_out, int out_size, void* d_ws,
                              size_t ws_size, hipStream_t stream) {
  const float* x        = (const float*)d_in[0];
  const float* w_qkv    = (const float*)d_in[1];
  const float* b_qkv    = (const float*)d_in[2];
  const float* w_proj   = (const float*)d_in[3];
  const float* b_proj   = (const float*)d_in[4];
  const float* rel      = (const float*)d_in[5];
  const float* pe_w     = (const float*)d_in[6];
  const float* pe_b     = (const float*)d_in[7];
  float* out = (float*)d_out;

  char* ws = (char*)d_ws;
  bf16* xb  = (bf16*)(ws + OFF_XB);
  bf16* wqb = (bf16*)(ws + OFF_WQB);
  bf16* wpb = (bf16*)(ws + OFF_WPB);
  bf16* qb  = (bf16*)(ws + OFF_QB);
  bf16* kb  = (bf16*)(ws + OFF_KB);
  bf16* vT  = (bf16*)(ws + OFF_VT);
  bf16* AO  = (bf16*)(ws + OFF_AO);

  cvt_bf16_kernel<<<20480, 256, 0, stream>>>(x, w_qkv, w_proj, xb, wqb, wpb);
  qkv_gemm_kernel<<<6144, 256, 0, stream>>>(xb, wqb, b_qkv, qb, kb, vT);

  (void)hipFuncSetAttribute(reinterpret_cast<const void*>(attn_kernel),
                            hipFuncAttributeMaxDynamicSharedMemorySize,
                            (int)SMEM_TOTAL);
  attn_kernel<<<2048, 256, SMEM_TOTAL, stream>>>(qb, kb, vT, rel, pe_w, pe_b, AO);

  proj_gemm_kernel<<<2048, 256, 0, stream>>>(AO, wpb, b_proj, out);
}
